// Agent_gcnlstm_53455162966160
// MI455X (gfx1250) — matrix operand order repaired and checked by reading, not compiled, not run on hardware
//
#include <hip/hip_runtime.h>

typedef __attribute__((ext_vector_type(16))) _Float16 v16h;
typedef __attribute__((ext_vector_type(8)))  _Float16 v8h;
typedef __attribute__((ext_vector_type(4)))  _Float16 v4h;
typedef __attribute__((ext_vector_type(8)))  float    v8f;

#define B_   8
#define L_   16
#define M_   256
#define H_   53
#define LB_  128
#define HID_ 64
#define SCLD 264

__device__ __forceinline__ float sigf(float x){ return 1.0f/(1.0f+__expf(-x)); }

__device__ __forceinline__ v8f wmma_f16(v16h a, v16h b, v8f c){
  return __builtin_amdgcn_wmma_f32_16x16x32_f16(false, a, false, b, (short)0, c, false, false);
}

// A fragment 16x32 f16 from row-major source, leading dim ld (elements).
// lane<16: elems 0-15 -> K0-15 ; lane>=16: K16-31.
// Vectorized: two aligned 16B loads per lane.
__device__ __forceinline__ v16h load_a(const _Float16* p, int ld){
  int lane = threadIdx.x & 31;
  const _Float16* r = p + (lane & 15) * ld + ((lane < 16) ? 0 : 16);
  union { v16h v; v8h h[2]; } a;
  a.h[0] = *(const v8h*)r;
  a.h[1] = *(const v8h*)(r + 8);
  return a.v;
}

// A fragment with only first 16 K valid (K16..31 zero) — head-dim-16 QK^T.
__device__ __forceinline__ v16h load_a_k16(const _Float16* p, int ld){
  int lane = threadIdx.x & 31;
  const _Float16* r = p + (lane & 15) * ld;
  union { v16h v; v8h h[2]; } a;
  v8h z = {};
  a.h[0] = (lane < 16) ? *(const v8h*)r : z;
  a.h[1] = (lane < 16) ? *(const v8h*)(r + 8) : z;
  return a.v;
}

// B fragment from pre-swizzled weights: fragment f, lane-contiguous 32B.
__device__ __forceinline__ v16h load_b_sw(const _Float16* base, int f){
  int lane = threadIdx.x & 31;
  return *(const v16h*)(base + (size_t)((f << 5) + lane) * 16);
}

// B fragment from N-major [N][K] source (row per output column), contiguous K.
__device__ __forceinline__ v16h load_b_rows(const _Float16* p, int ld){
  int lane = threadIdx.x & 31;
  const _Float16* r = p + (lane & 15) * ld + ((lane < 16) ? 0 : 16);
  return *(const v16h*)r;
}

// C frag store: lanes 0-15 -> rows 0-7, lanes 16-31 -> rows 8-15; column = lane&15.
__device__ __forceinline__ void store_c_h(v8f c, _Float16* p, int ld){
  int lane = threadIdx.x & 31;
  int col = lane & 15;
  int rb = (lane < 16) ? 0 : 8;
  #pragma unroll
  for (int v=0; v<8; ++v) p[(rb + v) * ld + col] = (_Float16)c[v];
}

__device__ __forceinline__ void store_c_f(v8f c, float* p, int ld){
  int lane = threadIdx.x & 31;
  int col = lane & 15;
  int rb = (lane < 16) ? 0 : 8;
  #pragma unroll
  for (int v=0; v<8; ++v) p[(rb + v) * ld + col] = c[v];
}

// ------------- weight packing: f32 -> f16 fragments in WMMA lane order -------------
// Layout per matrix: frag f -> 32 lanes x 16 halves contiguous (32B per lane).
// Frag (kk,nt): element j of lane: K = kk*32 + (lane<16?0:16) + j, N = nt*16 + (lane&15).
__global__ void pack_weights(const float* Wa, const float* Wg, const float* Qw, const float* Kw,
                             const float* W2, const float* Wia, const float* Wha,
                             const float* Wic, const float* Whc,
                             _Float16* WaT, _Float16* WgT, _Float16* QwH, _Float16* KwH,
                             _Float16* W2T, _Float16* WiaT, _Float16* WhaT,
                             _Float16* WicT, _Float16* WhcT){
  int idx = blockIdx.x * blockDim.x + threadIdx.x;   // < 16384
  int lane = (idx >> 4) & 31;
  int j = idx & 15;
  int kb = (lane < 16) ? 0 : 16;
  int col = lane & 15;
  if (idx < 4096){                      // K=64, N=64 : 8 frags (f = kk*4+nt)
    int f = idx >> 9;
    int kk = f >> 2, nt = f & 3;
    int k = kk*32 + kb + j;
    int n = nt*16 + col;
    WaT[idx] = (k < H_) ? (_Float16)Wa[n*H_ + k] : (_Float16)0.0f;
    WgT[idx] = (k < H_) ? (_Float16)Wg[n*H_ + k] : (_Float16)0.0f;
    QwH[idx] = (_Float16)Qw[k*64 + n];
    KwH[idx] = (_Float16)Kw[k*64 + n];
  }
  if (idx < 8192){                      // W2: K=128, N=64 : 16 frags (f = kk*4+nt)
    int f = idx >> 9;
    int kk = f >> 2, nt = f & 3;
    int k = kk*32 + kb + j;
    int n = nt*16 + col;
    W2T[idx] = (_Float16)W2[n*128 + k];
  }
  {                                     // Wih/Whh: K=64, N=256 : 32 frags (f = kk*16+nt)
    int f = idx >> 9;
    int kk = f >> 4, nt = f & 15;
    int k = kk*32 + kb + j;
    int n = nt*16 + col;
    WiaT[idx] = (_Float16)Wia[n*64 + k];
    WhaT[idx] = (_Float16)Wha[n*64 + k];
    WicT[idx] = (_Float16)Wic[n*64 + k];
    WhcT[idx] = (_Float16)Whc[n*64 + k];
  }
}

// x_actor [B,L,M,53] f32 -> xa [L*B, M, 64] f16 (padded), 4 cols per thread
__global__ void convert_x(const float* x, _Float16* xa){
  int idx = blockIdx.x * blockDim.x + threadIdx.x;   // < 128*256*16
  int c4 = (idx & 15) * 4;
  int m = (idx >> 4) & 255;
  int lb = idx >> 12;
  int l = lb >> 3, b = lb & 7;
  const float* src = x + ((size_t)(b*L_ + l)*M_ + m)*H_;
  v4h o;
  #pragma unroll
  for (int t=0;t<4;++t){
    int h = c4 + t;
    o[t] = (h < H_) ? (_Float16)src[h] : (_Float16)0.0f;
  }
  *(v4h*)(xa + ((size_t)lb*M_ + m)*64 + c4) = o;
}

// ------------- projections: a_lin, s (+ transposed sT), q, k : 1 wave / 16-row tile ----
__global__ void proj_kernel(const _Float16* xa, const _Float16* WaT, const _Float16* WgT,
                            const _Float16* QwH, const _Float16* KwH,
                            const float* ba, const float* bg,
                            _Float16* a_h, _Float16* s_h, _Float16* sT,
                            _Float16* q_h, _Float16* k_h){
  __shared__ __align__(32) _Float16 s_lds[16][64];
  int lb = blockIdx.x >> 4;
  int mt = blockIdx.x & 15;
  int lane = threadIdx.x & 31;
  int col = lane & 15;
  int rb = (lane < 16) ? 0 : 8;
  const _Float16* xt = xa + ((size_t)lb*M_ + mt*16)*64;
  v16h ax0 = load_a(xt, 64);
  v16h ax1 = load_a(xt + 32, 64);
  // s = x @ WgT + bg (LDS + row-major global + transposed global)
  #pragma unroll
  for (int nt=0; nt<4; ++nt){
    v8f acc = {};
    acc = wmma_f16(ax0, load_b_sw(WgT, nt), acc);
    acc = wmma_f16(ax1, load_b_sw(WgT, 4 + nt), acc);
    float bv = bg[nt*16 + col];
    #pragma unroll
    for (int v=0; v<8; ++v) acc[v] += bv;
    store_c_h(acc, &s_lds[0][nt*16], 64);
    store_c_h(acc, s_h + ((size_t)lb*M_ + mt*16)*64 + nt*16, 64);
    v8h tv;                                    // transposed: channel-major, m contiguous
    #pragma unroll
    for (int v=0; v<8; ++v) tv[v] = (_Float16)acc[v];
    *(v8h*)(sT + ((size_t)lb*64 + nt*16 + col)*M_ + mt*16 + rb) = tv;
  }
  // a_lin = x @ WaT + ba
  #pragma unroll
  for (int nt=0; nt<4; ++nt){
    v8f acc = {};
    acc = wmma_f16(ax0, load_b_sw(WaT, nt), acc);
    acc = wmma_f16(ax1, load_b_sw(WaT, 4 + nt), acc);
    float bv = ba[nt*16 + col];
    #pragma unroll
    for (int v=0; v<8; ++v) acc[v] += bv;
    store_c_h(acc, a_h + ((size_t)lb*M_ + mt*16)*64 + nt*16, 64);
  }
  __syncthreads();
  v16h as0 = load_a(&s_lds[0][0], 64);
  v16h as1 = load_a(&s_lds[0][32], 64);
  #pragma unroll
  for (int nt=0; nt<4; ++nt){
    v8f acc = {};
    acc = wmma_f16(as0, load_b_sw(QwH, nt), acc);
    acc = wmma_f16(as1, load_b_sw(QwH, 4 + nt), acc);
    store_c_h(acc, q_h + ((size_t)lb*M_ + mt*16)*64 + nt*16, 64);
  }
  #pragma unroll
  for (int nt=0; nt<4; ++nt){
    v8f acc = {};
    acc = wmma_f16(as0, load_b_sw(KwH, nt), acc);
    acc = wmma_f16(as1, load_b_sw(KwH, 4 + nt), acc);
    store_c_h(acc, k_h + ((size_t)lb*M_ + mt*16)*64 + nt*16, 64);
  }
}

// ------------- flash-style masked attention (4 waves = 4 heads per block) -------------
__global__ void attn_kernel(const _Float16* q_h, const _Float16* k_h, const _Float16* sT,
                            const int* G, _Float16* so_h){
  __shared__ __align__(32) _Float16 sc[4][16][SCLD];
  int qt = blockIdx.x;       // 16 query tiles
  int lb = blockIdx.y;       // 128 batch-steps
  int w = threadIdx.x >> 5;  // head
  int lane = threadIdx.x & 31;
  int col = lane & 15;
  int rb = (lane < 16) ? 0 : 8;
  // scores = Q K^T / sqrt(128), masked by Gmat, into LDS f16
  v16h aq = load_a_k16(q_h + ((size_t)lb*M_ + qt*16)*64 + w*16, 64);
  for (int st=0; st<16; ++st){
    v16h bk = {};
    if (lane < 16)
      bk = *(const v16h*)(k_h + ((size_t)lb*M_ + st*16 + lane)*64 + w*16);
    v8f acc = {};
    acc = wmma_f16(aq, bk, acc);
    #pragma unroll
    for (int v=0; v<8; ++v){
      int lm = qt*16 + rb + v;
      int sm = st*16 + col;
      float sv = acc[v] * 0.08838834764831845f;     // 1/sqrt(128)
      if (G[lm*M_ + sm] <= 0) sv = -30000.0f;
      sc[w][rb+v][sm] = (_Float16)sv;
    }
  }
  __syncthreads();
  // row softmax over 256 keys, vectorized LDS scans (lane r -> row r)
  if (lane < 16){
    _Float16* rowp = &sc[w][lane][0];
    float mx = -1e30f;
    for (int jb=0; jb<32; ++jb){
      v8h c = *(const v8h*)(rowp + jb*8);
      #pragma unroll
      for (int t=0;t<8;++t) mx = fmaxf(mx, (float)c[t]);
    }
    float sum = 0.0f;
    for (int jb=0; jb<32; ++jb){
      v8h c = *(const v8h*)(rowp + jb*8);
      #pragma unroll
      for (int t=0;t<8;++t) sum += __expf((float)c[t] - mx);
    }
    float inv = 1.0f / sum;
    for (int jb=0; jb<32; ++jb){
      v8h c = *(const v8h*)(rowp + jb*8);
      v8h o;
      #pragma unroll
      for (int t=0;t<8;++t) o[t] = (_Float16)(__expf((float)c[t] - mx) * inv);
      *(v8h*)(rowp + jb*8) = o;
    }
  }
  __syncthreads();
  // s_out = att @ V ; V from transposed s (contiguous 32B per lane)
  v8f acc = {};
  #pragma unroll
  for (int kk=0; kk<8; ++kk){
    v16h aA = load_a(&sc[w][0][kk*32], SCLD);
    v16h bV = load_b_rows(sT + ((size_t)lb*64 + w*16)*M_ + kk*32, M_);
    acc = wmma_f16(aA, bV, acc);
  }
  store_c_h(acc, so_h + ((size_t)lb*M_ + qt*16)*64 + w*16, 64);
}

// ------------- c_feat = cat(s, s_out) @ W2^T + b2 -------------
__global__ void cfeat_kernel(const _Float16* s_h, const _Float16* so_h, const _Float16* W2T,
                             const float* b2, _Float16* cf_h){
  int lb = blockIdx.x >> 4;
  int mt = blockIdx.x & 15;
  int col = threadIdx.x & 15;
  const _Float16* st = s_h  + ((size_t)lb*M_ + mt*16)*64;
  const _Float16* ot = so_h + ((size_t)lb*M_ + mt*16)*64;
  v16h a0 = load_a(st, 64);
  v16h a1 = load_a(st + 32, 64);
  v16h a2 = load_a(ot, 64);
  v16h a3 = load_a(ot + 32, 64);
  #pragma unroll
  for (int nt=0; nt<4; ++nt){
    v8f acc = {};
    acc = wmma_f16(a0, load_b_sw(W2T, 0*4 + nt), acc);
    acc = wmma_f16(a1, load_b_sw(W2T, 1*4 + nt), acc);
    acc = wmma_f16(a2, load_b_sw(W2T, 2*4 + nt), acc);
    acc = wmma_f16(a3, load_b_sw(W2T, 3*4 + nt), acc);
    float bv = b2[nt*16 + col];
    #pragma unroll
    for (int v=0; v<8; ++v) acc[v] += bv;
    store_c_h(acc, cf_h + ((size_t)lb*M_ + mt*16)*64 + nt*16, 64);
  }
}

// ------------- LSTM: 16 seqs / block; wave w = gate w; weights register-resident -------
__global__ void lstm_kernel(const _Float16* xin, const _Float16* WiT, const _Float16* WhT,
                            const float* bi, const float* bh, float* hout){
  __shared__ float    gates[16][256];
  __shared__ __align__(32) _Float16 hs[16][64];
  __shared__ float    cs[16][64];
  int nt = blockIdx.x;           // seq tile; n = nt*16 + r  (n = b*M + m)
  int b  = (nt*16) >> 8;
  int m0 = (nt*16) & 255;
  int w  = threadIdx.x >> 5;     // gate: 0=i 1=f 2=g 3=o (cols w*64..)
  int lane = threadIdx.x & 31;
  int col = lane & 15;
  for (int i = threadIdx.x; i < 16*64; i += blockDim.x){
    hs[i>>6][i&63] = (_Float16)0.0f;
    cs[i>>6][i&63] = 0.0f;
  }
  // preload this gate's weight fragments into VGPRs (reused across 16 steps)
  v16h wi[2][4], wh[2][4];
  #pragma unroll
  for (int kk=0; kk<2; ++kk)
    #pragma unroll
    for (int nn=0; nn<4; ++nn){
      wi[kk][nn] = load_b_sw(WiT, kk*16 + w*4 + nn);
      wh[kk][nn] = load_b_sw(WhT, kk*16 + w*4 + nn);
    }
  float bsum[4];
  #pragma unroll
  for (int nn=0; nn<4; ++nn){
    int n0 = w*64 + nn*16 + col;
    bsum[nn] = bi[n0] + bh[n0];
  }
  __syncthreads();
  for (int l = 0; l < L_; ++l){
    const _Float16* xt = xin + ((size_t)(l*B_ + b)*M_ + m0)*64;
    v16h ax0 = load_a(xt, 64);
    v16h ax1 = load_a(xt + 32, 64);
    v16h ah0 = load_a(&hs[0][0], 64);
    v16h ah1 = load_a(&hs[0][32], 64);
    #pragma unroll
    for (int nn=0; nn<4; ++nn){
      v8f acc = {};
      acc = wmma_f16(ax0, wi[0][nn], acc);
      acc = wmma_f16(ax1, wi[1][nn], acc);
      acc = wmma_f16(ah0, wh[0][nn], acc);
      acc = wmma_f16(ah1, wh[1][nn], acc);
      #pragma unroll
      for (int v=0; v<8; ++v) acc[v] += bsum[nn];
      store_c_f(acc, &gates[0][w*64 + nn*16], 256);
    }
    __syncthreads();
    for (int i = threadIdx.x; i < 16*64; i += blockDim.x){
      int r = i >> 6, j = i & 63;
      float gi = gates[r][j];
      float gf = gates[r][64 + j];
      float gg = gates[r][128 + j];
      float go = gates[r][192 + j];
      float c = sigf(gf)*cs[r][j] + sigf(gi)*tanhf(gg);
      float h = sigf(go)*tanhf(c);
      cs[r][j] = c;
      hs[r][j] = (_Float16)h;
      if (l == L_-1) hout[(size_t)(nt*16 + r)*64 + j] = h;
    }
    __syncthreads();
  }
}

// ------------- tiny heads: logits/logp/entropy + value -------------
__global__ void heads_kernel(const float* ah, const float* ch, const int* action,
                             const float* Wma1, const float* bma1,
                             const float* Wma2, const float* bma2,
                             const float* Wmc1, const float* bmc1,
                             const float* Wmc2, const float* bmc2,
                             float* out){
  int n = blockIdx.x * blockDim.x + threadIdx.x;
  if (n >= B_*M_) return;
  float hid[64];
  const float* a = ah + (size_t)n*64;
  for (int j=0;j<64;++j){
    float s = bma1[j];
    for (int k=0;k<64;++k) s += a[k] * Wma1[j*64 + k];
    hid[j] = tanhf(s);
  }
  float logits[15];
  for (int t=0;t<15;++t){
    float s = bma2[t];
    for (int k=0;k<64;++k) s += hid[k] * Wma2[t*64 + k];
    logits[t] = s;
  }
  float lp_sum = 0.0f, ent_sum = 0.0f;
  for (int g=0; g<3; ++g){
    float mx = -1e30f;
    for (int c=0;c<5;++c) mx = fmaxf(mx, logits[g*5+c]);
    float se = 0.0f;
    for (int c=0;c<5;++c) se += __expf(logits[g*5+c] - mx);
    float lse = mx + __logf(se);
    int act = action[n*3 + g];
    lp_sum += logits[g*5 + act] - lse;
    for (int c=0;c<5;++c){
      float lpv = logits[g*5+c] - lse;
      ent_sum -= __expf(lpv) * lpv;
    }
    out[n*3 + g] = (float)act;
  }
  const float* cc = ch + (size_t)n*64;
  float val = bmc2[0];
  for (int j=0;j<64;++j){
    float s = bmc1[j];
    for (int k=0;k<64;++k) s += cc[k] * Wmc1[j*64 + k];
    val += tanhf(s) * Wmc2[j];
  }
  out[6144  + n] = lp_sum;
  out[8192  + n] = ent_sum;
  out[10240 + n] = val;
}

extern "C" void kernel_launch(void* const* d_in, const int* in_sizes, int n_in,
                              void* d_out, int out_size, void* d_ws, size_t ws_size,
                              hipStream_t stream){
  (void)in_sizes; (void)n_in; (void)out_size; (void)ws_size;
  const float* x_actor = (const float*)d_in[0];
  const int*   action  = (const int*)d_in[2];
  const int*   Gmat    = (const int*)d_in[3];
  const float* Wa = (const float*)d_in[4];   const float* ba = (const float*)d_in[5];
  const float* Wg = (const float*)d_in[6];   const float* bg = (const float*)d_in[7];
  const float* Qw = (const float*)d_in[8];   const float* Kw = (const float*)d_in[9];
  const float* W2 = (const float*)d_in[10];  const float* b2 = (const float*)d_in[11];
  const float* Wih_a = (const float*)d_in[12]; const float* Whh_a = (const float*)d_in[13];
  const float* bih_a = (const float*)d_in[14]; const float* bhh_a = (const float*)d_in[15];
  const float* Wih_c = (const float*)d_in[16]; const float* Whh_c = (const float*)d_in[17];
  const float* bih_c = (const float*)d_in[18]; const float* bhh_c = (const float*)d_in[19];
  const float* Wma1 = (const float*)d_in[20]; const float* bma1 = (const float*)d_in[21];
  const float* Wma2 = (const float*)d_in[22]; const float* bma2 = (const float*)d_in[23];
  const float* Wmc1 = (const float*)d_in[24]; const float* bmc1 = (const float*)d_in[25];
  const float* Wmc2 = (const float*)d_in[26]; const float* bmc2 = (const float*)d_in[27];
  float* out = (float*)d_out;

  _Float16* p = (_Float16*)d_ws;
  const size_t T = (size_t)LB_ * M_ * 64;   // 2,097,152 halves per tensor
  _Float16* xa_h = p;  p += T;
  _Float16* a_h  = p;  p += T;
  _Float16* s_h  = p;  p += T;
  _Float16* sT_h = p;  p += T;
  _Float16* q_h  = p;  p += T;
  _Float16* k_h  = p;  p += T;
  _Float16* so_h = p;  p += T;
  _Float16* cf_h = p;  p += T;
  _Float16* WaT  = p;  p += 4096;
  _Float16* WgT  = p;  p += 4096;
  _Float16* QwH  = p;  p += 4096;
  _Float16* KwH  = p;  p += 4096;
  _Float16* W2T  = p;  p += 8192;
  _Float16* WiaT = p;  p += 16384;
  _Float16* WhaT = p;  p += 16384;
  _Float16* WicT = p;  p += 16384;
  _Float16* WhcT = p;  p += 16384;
  float* ahF = (float*)p;
  float* chF = ahF + (size_t)B_ * M_ * 64;

  pack_weights<<<64, 256, 0, stream>>>(Wa, Wg, Qw, Kw, W2, Wih_a, Whh_a, Wih_c, Whh_c,
                                       WaT, WgT, QwH, KwH, W2T, WiaT, WhaT, WicT, WhcT);
  convert_x<<<2048, 256, 0, stream>>>(x_actor, xa_h);
  proj_kernel<<<2048, 32, 0, stream>>>(xa_h, WaT, WgT, QwH, KwH, ba, bg,
                                       a_h, s_h, sT_h, q_h, k_h);
  attn_kernel<<<dim3(16,128), 128, 0, stream>>>(q_h, k_h, sT_h, Gmat, so_h);
  cfeat_kernel<<<2048, 32, 0, stream>>>(s_h, so_h, W2T, b2, cf_h);
  lstm_kernel<<<128, 128, 0, stream>>>(a_h,  WiaT, WhaT, bih_a, bhh_a, ahF);
  lstm_kernel<<<128, 128, 0, stream>>>(cf_h, WicT, WhcT, bih_c, bhh_c, chF);
  heads_kernel<<<8, 256, 0, stream>>>(ahF, chF, action, Wma1, bma1, Wma2, bma2,
                                      Wmc1, bmc1, Wmc2, bmc2, out);
}
